// CMF_Block_23845658427435
// MI455X (gfx1250) — matrix operand order repaired and checked by reading, not compiled, not run on hardware
//
#include <hip/hip_runtime.h>
#include <math.h>

typedef __attribute__((ext_vector_type(16))) __bf16        v16bf;
typedef __attribute__((ext_vector_type(8)))  float         v8f;
typedef __attribute__((ext_vector_type(8)))  unsigned int  v8u;

#define WMMA_BF16(A, B, C) \
  __builtin_amdgcn_wmma_f32_16x16x32_bf16(false, (A), false, (B), (short)0, (C), false, false)

// ---- CDNA5 async global->LDS staging (ASYNCcnt path), with safe fallback ----
#if defined(__AMDGCN__) && __has_builtin(__builtin_amdgcn_global_load_async_to_lds_b128)
#define HAVE_ASYNC 1
#else
#define HAVE_ASYNC 0
#endif

typedef int v4i_g __attribute__((__vector_size__(16)));
typedef __attribute__((address_space(1))) v4i_g as1_v4i;
typedef __attribute__((address_space(3))) v4i_g as3_v4i;

__device__ __forceinline__ void cp_async16(void* lds_dst, const void* gsrc) {
#if HAVE_ASYNC
  __builtin_amdgcn_global_load_async_to_lds_b128(
      (as1_v4i*)(unsigned long long)gsrc,
      (as3_v4i*)(unsigned int)(unsigned long long)lds_dst,
      0, 0);
#else
  *(uint4*)lds_dst = *(const uint4*)gsrc;
#endif
}

__device__ __forceinline__ void async_wait0() {
#if HAVE_ASYNC
#if __has_builtin(__builtin_amdgcn_s_wait_asynccnt)
  __builtin_amdgcn_s_wait_asynccnt(0);
#else
  asm volatile("s_wait_asynccnt 0x0" ::: "memory");
#endif
#endif
}

__device__ __forceinline__ unsigned pk_bf16(float lo, float hi) {
  union { float f; unsigned u; } a, b;
  a.f = lo; b.f = hi;
  unsigned ul = (a.u + 0x7FFFu + ((a.u >> 16) & 1u)) >> 16;
  unsigned uh = (b.u + 0x7FFFu + ((b.u >> 16) & 1u)) >> 16;
  return (ul & 0xFFFFu) | (uh << 16);
}
__device__ __forceinline__ unsigned short bf16_1(float x) {
  union { float f; unsigned u; } a; a.f = x;
  return (unsigned short)((a.u + 0x7FFFu + ((a.u >> 16) & 1u)) >> 16);
}

// ---------------------------------------------------------------------------
// Kernel 0: one-shot w4 f32 -> bf16 conversion (65536 elements).
// ---------------------------------------------------------------------------
__global__ __launch_bounds__(256) void w4cvt_kernel(
    const float* __restrict__ w4, unsigned int* __restrict__ w4bf)
{
  int base = (blockIdx.x * 256 + threadIdx.x) * 4;
  for (int p = 0; p < 4; ++p) {
    int pi = base + p;                 // pair index, 0..32767
    w4bf[pi] = pk_bf16(w4[2 * pi], w4[2 * pi + 1]);
  }
}

// ---------------------------------------------------------------------------
// Kernel 1: grouped 1x1 convs (q from rgb; k,v from ir), f32 compute.
// Writes qT, kT as [b][n][c] bf16 and v as [b][c][n] bf16.
// grid (64 i-tiles, 4 groups, 4 batches), 256 threads.
// ---------------------------------------------------------------------------
__global__ __launch_bounds__(256) void qkv_kernel(
    const float* __restrict__ rgb, const float* __restrict__ ir,
    const float* __restrict__ wq, const float* __restrict__ bq,
    const float* __restrict__ wk, const float* __restrict__ bk,
    const float* __restrict__ wv, const float* __restrict__ bv,
    unsigned short* __restrict__ qT, unsigned short* __restrict__ kT,
    unsigned short* __restrict__ vP)
{
  __shared__ __align__(16) float sWq[64 * 64];
  __shared__ __align__(16) float sWk[64 * 64];
  __shared__ __align__(16) float sWv[64 * 64];
  __shared__ __align__(16) float sXr[64 * 64];
  __shared__ __align__(16) float sXi[64 * 64];

  const int t  = threadIdx.x;
  const int i0 = blockIdx.x * 64;
  const int g  = blockIdx.y;
  const int b  = blockIdx.z;

  const float* wq_g = wq + g * 4096;
  const float* wk_g = wk + g * 4096;
  const float* wv_g = wv + g * 4096;
  for (int e = t; e < 1024; e += 256) {     // 16B chunks (4 f32)
    cp_async16(sWq + e * 4, wq_g + e * 4);
    cp_async16(sWk + e * 4, wk_g + e * 4);
    cp_async16(sWv + e * 4, wv_g + e * 4);
    int ci = e >> 4, col4 = e & 15;
    int src = (b * 256 + g * 64 + ci) * 4096 + i0 + col4 * 4;
    cp_async16(sXr + ci * 64 + col4 * 4, rgb + src);
    cp_async16(sXi + ci * 64 + col4 * 4, ir + src);
  }
  async_wait0();
  __syncthreads();

  const int i   = t & 63;   // position within tile
  const int cob = t >> 6;   // 0..3

  float aq[16], ak[16], av[16];
  for (int u = 0; u < 16; ++u) { aq[u] = 0.f; ak[u] = 0.f; av[u] = 0.f; }

  for (int ci = 0; ci < 64; ++ci) {
    float xr = sXr[ci * 64 + i];
    float xi = sXi[ci * 64 + i];
    for (int u = 0; u < 16; ++u) {
      int co = cob + 4 * u;
      aq[u] = fmaf(sWq[co * 64 + ci], xr, aq[u]);
      ak[u] = fmaf(sWk[co * 64 + ci], xi, ak[u]);
      av[u] = fmaf(sWv[co * 64 + ci], xi, av[u]);
    }
  }

  const int gi = i0 + i;
  for (int u = 0; u < 16; ++u) {
    int co = cob + 4 * u;
    int c  = g * 64 + co;
    qT[(b * 4096 + gi) * 256 + c] = bf16_1(aq[u] + bq[c]);
    kT[(b * 4096 + gi) * 256 + c] = bf16_1(ak[u] + bk[c]);
    vP[(b * 256 + c) * 4096 + gi] = bf16_1(av[u] + bv[c]);
  }
}

// ---------------------------------------------------------------------------
// Kernel 2: flash attention, BN=128 j-chunks. WG = 8 waves, 32 query rows,
// online softmax. Each wave: 2 S tiles (32 WMMA/chunk). K and V chunks
// time-share one ~68KB LDS buffer, staged via async global->LDS.
// grid (128 i-tiles, 4 batches), 256 threads.
// ---------------------------------------------------------------------------
#define KSTR 528   // 128 rows x (256 c * 2B), +16 pad: 16B aligned, 4-bank shift
#define VSTR 272   // 256 rows x (128 j * 2B), +16 pad
#define PSTR 264   // 32 rows  x (128 j * 2B), +8 pad

__global__ __launch_bounds__(256) void attn_kernel(
    const unsigned short* __restrict__ qT,
    const unsigned short* __restrict__ kT,
    const unsigned short* __restrict__ vP,
    unsigned short* __restrict__ zT)
{
  __shared__ __align__(16) unsigned char kv[256 * VSTR];   // K then V (reused)
  __shared__ __align__(16) unsigned char pls[32 * PSTR];   // P (bf16)
  __shared__ float pmax[128];   // [q4][row 0..31]
  __shared__ float psum[128];

  const int t    = threadIdx.x;
  const int wv_  = t >> 5;
  const int lane = t & 31;
  const int hf   = lane >> 4;
  const int l16  = lane & 15;
  const int iw   = wv_ >> 2;   // which 16-row block (0,1)
  const int q4   = wv_ & 3;    // j-subtile pair in S phase / c-block in Z phase
  const int b    = blockIdx.y;
  const int i0   = blockIdx.x * 32;

  const unsigned short* qb = qT + (size_t)b * 4096 * 256;
  const unsigned short* kb = kT + (size_t)b * 4096 * 256;
  const unsigned short* vb = vP + (size_t)b * 256 * 4096;

  // Persistent Q fragments: A-matrix rows i0+iw*16 .. +15, all 256 c.
  v16bf qa[8];
  {
    const unsigned short* qrow = qb + (i0 + iw * 16 + l16) * 256;
    for (int kc = 0; kc < 8; ++kc) {
      v8u tmp;
      for (int vvi = 0; vvi < 8; ++vvi) {
        int c = kc * 32 + hf * 16 + 2 * vvi;
        tmp[vvi] = *(const unsigned int*)(qrow + c);
      }
      qa[kc] = __builtin_bit_cast(v16bf, tmp);
    }
  }

  v8f acc[4] = {};
  float mprev[8], lrun[8];
  for (int r = 0; r < 8; ++r) { mprev[r] = -1e30f; lrun[r] = 0.f; }

  for (int j0 = 0; j0 < 4096; j0 += 128) {
    // ---- stage K chunk: rows j0..j0+127, 256 c each (bf16) ----
    for (int e = t; e < 4096; e += 256) {
      int row = e >> 5, col = e & 31;
      cp_async16(kv + row * KSTR + col * 16, kb + (j0 + row) * 256 + col * 8);
      if (j0 + 128 < 4096)
        __builtin_prefetch(kb + (j0 + 128 + row) * 256 + col * 8, 0, 0);
    }
    async_wait0();
    __syncthreads();

    // ---- two S tiles [16 i x 16 j] per wave (j-subtiles q4 and q4+4) ----
    v8f s0 = {}, s1 = {};
    {
      int jr0 = q4 * 16 + l16;
      int jr1 = jr0 + 64;
      for (int kc = 0; kc < 8; ++kc) {
        v8u tb0, tb1;
        for (int vvi = 0; vvi < 8; ++vvi) {
          int c = kc * 32 + hf * 16 + 2 * vvi;
          tb0[vvi] = *(const unsigned int*)(kv + jr0 * KSTR + c * 2);
          tb1[vvi] = *(const unsigned int*)(kv + jr1 * KSTR + c * 2);
        }
        s0 = WMMA_BF16(qa[kc], __builtin_bit_cast(v16bf, tb0), s0);
        s1 = WMMA_BF16(qa[kc], __builtin_bit_cast(v16bf, tb1), s1);
      }
    }
    // scale by c^-0.5 = 1/16; fused row max over both tiles, then lanes
    float lm[8];
    for (int r = 0; r < 8; ++r) {
      s0[r] *= 0.0625f;
      s1[r] *= 0.0625f;
      lm[r] = fmaxf(s0[r], s1[r]);
    }
    for (int m = 1; m < 16; m <<= 1)
      for (int r = 0; r < 8; ++r)
        lm[r] = fmaxf(lm[r], __shfl_xor(lm[r], m, 32));
    if (l16 == 0)
      for (int r = 0; r < 8; ++r)
        pmax[q4 * 32 + iw * 16 + hf * 8 + r] = lm[r];
    __syncthreads();

    // ---- global row max, exp, row sums, P->bf16 to LDS ----
    float mnew[8], ls[8];
    for (int r = 0; r < 8; ++r) {
      int row = iw * 16 + hf * 8 + r;
      float mn = mprev[r];
      mn = fmaxf(mn, pmax[row]);
      mn = fmaxf(mn, pmax[32 + row]);
      mn = fmaxf(mn, pmax[64 + row]);
      mn = fmaxf(mn, pmax[96 + row]);
      mnew[r] = mn;
    }
    v8f p0, p1;
    for (int r = 0; r < 8; ++r) {
      p0[r] = __expf(s0[r] - mnew[r]);
      p1[r] = __expf(s1[r] - mnew[r]);
      ls[r] = p0[r] + p1[r];
    }
    for (int m = 1; m < 16; m <<= 1)
      for (int r = 0; r < 8; ++r)
        ls[r] += __shfl_xor(ls[r], m, 32);
    if (l16 == 0)
      for (int r = 0; r < 8; ++r)
        psum[q4 * 32 + iw * 16 + hf * 8 + r] = ls[r];
    for (int r = 0; r < 8; ++r) {
      int row = iw * 16 + hf * 8 + r;
      *(unsigned short*)(pls + row * PSTR + (q4 * 16 + l16) * 2)      = bf16_1(p0[r]);
      *(unsigned short*)(pls + row * PSTR + (64 + q4 * 16 + l16) * 2) = bf16_1(p1[r]);
    }
    __syncthreads();

    // ---- stage V chunk into same buffer: v[c][j0..j0+127] ----
    for (int e = t; e < 4096; e += 256) {
      int row = e >> 4, col = e & 15;
      cp_async16(kv + row * VSTR + col * 16, vb + row * 4096 + j0 + col * 8);
      if (j0 + 128 < 4096)
        __builtin_prefetch(vb + row * 4096 + j0 + 128 + col * 8, 0, 0);
    }
    async_wait0();
    __syncthreads();

    // ---- online rescale + Z += P @ V^T ----
    for (int r = 0; r < 8; ++r) {
      int row = iw * 16 + hf * 8 + r;
      float sum = psum[row] + psum[32 + row] + psum[64 + row] + psum[96 + row];
      float f = __expf(mprev[r] - mnew[r]);
      lrun[r] = lrun[r] * f + sum;
      mprev[r] = mnew[r];
      for (int cs = 0; cs < 4; ++cs) acc[cs][r] *= f;
    }
    for (int ks = 0; ks < 4; ++ks) {
      v8u ta;
      int prow = iw * 16 + l16;
      for (int vvi = 0; vvi < 8; ++vvi) {
        int kj = ks * 32 + hf * 16 + 2 * vvi;
        ta[vvi] = *(const unsigned int*)(pls + prow * PSTR + kj * 2);
      }
      v16bf pf = __builtin_bit_cast(v16bf, ta);
      for (int cs = 0; cs < 4; ++cs) {
        v8u tb;
        int c = q4 * 64 + cs * 16 + l16;
        for (int vvi = 0; vvi < 8; ++vvi) {
          int kj = ks * 32 + hf * 16 + 2 * vvi;
          tb[vvi] = *(const unsigned int*)(kv + c * VSTR + kj * 2);
        }
        acc[cs] = WMMA_BF16(pf, __builtin_bit_cast(v16bf, tb), acc[cs]);
      }
    }
    __syncthreads();
  }

  // ---- finalize: z /= l, store zT [b][i][c] bf16 ----
  unsigned short* zb = zT + (size_t)b * 4096 * 256;
  for (int cs = 0; cs < 4; ++cs) {
    int c = q4 * 64 + cs * 16 + l16;
    for (int r = 0; r < 8; ++r) {
      int gi = i0 + iw * 16 + hf * 8 + r;
      zb[gi * 256 + c] = bf16_1(acc[cs][r] / lrun[r]);
    }
  }
}

// ---------------------------------------------------------------------------
// Kernel 3: full 1x1 conv (w4, pre-converted bf16) via WMMA + BN(eval) +
// Mish + residual. grid (128 i-tiles, 4 batches), 256 threads (8 waves).
// ---------------------------------------------------------------------------
#define ZSTR 528

__global__ __launch_bounds__(256) void conv4_kernel(
    const unsigned short* __restrict__ zT,
    const unsigned short* __restrict__ w4bf, const float* __restrict__ b4,
    const float* __restrict__ gamma, const float* __restrict__ beta,
    const float* __restrict__ rmean, const float* __restrict__ rvar,
    const float* __restrict__ rgb, float* __restrict__ out)
{
  __shared__ __align__(16) unsigned char zlds[32 * ZSTR];
  __shared__ float tr[8][16][17];

  const int t    = threadIdx.x;
  const int wv_  = t >> 5;
  const int lane = t & 31;
  const int hf   = lane >> 4;
  const int l16  = lane & 15;
  const int iw   = wv_ & 1;    // 16-row block
  const int og   = wv_ >> 1;   // 64-wide o block
  const int b    = blockIdx.y;
  const int i0   = blockIdx.x * 32;

  // stage zT tile [32 i][256 c] bf16 via async copy
  for (int e = t; e < 1024; e += 256) {
    int row = e >> 5, col = e & 31;
    cp_async16(zlds + row * ZSTR + col * 16,
               zT + (size_t)b * 4096 * 256 + (i0 + row) * 256 + col * 8);
  }
  async_wait0();
  __syncthreads();

  v8f acc[4] = {};
  for (int kc = 0; kc < 8; ++kc) {
    v8u ta;
    int zr = iw * 16 + l16;
    for (int vvi = 0; vvi < 8; ++vvi) {
      int c = kc * 32 + hf * 16 + 2 * vvi;
      ta[vvi] = *(const unsigned int*)(zlds + zr * ZSTR + c * 2);
    }
    v16bf af = __builtin_bit_cast(v16bf, ta);
    for (int s = 0; s < 4; ++s) {
      v8u tb;
      int o = og * 64 + s * 16 + l16;
      const unsigned short* wrow = w4bf + o * 256;
      for (int vvi = 0; vvi < 8; ++vvi) {
        int c = kc * 32 + hf * 16 + 2 * vvi;
        tb[vvi] = *(const unsigned int*)(wrow + c);
      }
      acc[s] = WMMA_BF16(af, __builtin_bit_cast(v16bf, tb), acc[s]);
    }
  }

  for (int s = 0; s < 4; ++s) {
    int o = og * 64 + s * 16 + l16;
    float inv = gamma[o] * rsqrtf(rvar[o] + 1e-5f);
    float sh  = beta[o] - rmean[o] * inv;
    float bb  = b4[o];
    for (int r = 0; r < 8; ++r) {
      float y  = (acc[s][r] + bb) * inv + sh;
      float sp = (y > 20.f) ? y : log1pf(__expf(y));
      tr[wv_][hf * 8 + r][l16] = y * tanhf(sp);
    }
    __syncthreads();
    for (int r2 = 0; r2 < 8; ++r2) {
      int e  = lane + 32 * r2;
      int ol = e >> 4, il = e & 15;
      float mv = tr[wv_][il][ol];
      size_t gi = (size_t)(b * 256 + og * 64 + s * 16 + ol) * 4096
                + (size_t)(i0 + iw * 16 + il);
      out[gi] = rgb[gi] + mv;
    }
    __syncthreads();
  }
}

// ---------------------------------------------------------------------------
extern "C" void kernel_launch(void* const* d_in, const int* in_sizes, int n_in,
                              void* d_out, int out_size, void* d_ws, size_t ws_size,
                              hipStream_t stream) {
  const float* rgb   = (const float*)d_in[0];
  const float* ir    = (const float*)d_in[1];
  const float* w_q   = (const float*)d_in[2];
  const float* b_q   = (const float*)d_in[3];
  const float* w_k   = (const float*)d_in[4];
  const float* b_k   = (const float*)d_in[5];
  const float* w_v   = (const float*)d_in[6];
  const float* b_v   = (const float*)d_in[7];
  const float* w4    = (const float*)d_in[8];
  const float* b4    = (const float*)d_in[9];
  const float* gamma = (const float*)d_in[10];
  const float* beta  = (const float*)d_in[11];
  const float* rmean = (const float*)d_in[12];
  const float* rvar  = (const float*)d_in[13];
  float* out = (float*)d_out;

  unsigned char* ws = (unsigned char*)d_ws;
  unsigned short* qT   = (unsigned short*)(ws);
  unsigned short* kT   = (unsigned short*)(ws + (size_t)8 * 1024 * 1024);
  unsigned short* vP   = (unsigned short*)(ws + (size_t)16 * 1024 * 1024);
  unsigned short* zT   = (unsigned short*)(ws + (size_t)24 * 1024 * 1024);
  unsigned short* w4bf = (unsigned short*)(ws + (size_t)32 * 1024 * 1024);

  w4cvt_kernel<<<dim3(32), 256, 0, stream>>>(w4, (unsigned int*)w4bf);
  qkv_kernel<<<dim3(64, 4, 4), 256, 0, stream>>>(
      rgb, ir, w_q, b_q, w_k, b_k, w_v, b_v, qT, kT, vP);
  attn_kernel<<<dim3(128, 4), 256, 0, stream>>>(qT, kT, vP, zT);
  conv4_kernel<<<dim3(128, 4), 256, 0, stream>>>(
      zT, w4bf, b4, gamma, beta, rmean, rvar, rgb, out);
}
